// SimpleAttention_31095563223601
// MI455X (gfx1250) — matrix operand order repaired and checked by reading, not compiled, not run on hardware
//
#include <hip/hip_runtime.h>
#include <math.h>

// ---------- vector types ----------
typedef __attribute__((ext_vector_type(16))) __bf16 v16bf;
typedef __attribute__((ext_vector_type(8)))  float  v8f;
typedef __attribute__((ext_vector_type(4)))  unsigned int u32x4;
typedef __attribute__((ext_vector_type(4)))  int    i32x4;
typedef __attribute__((ext_vector_type(4)))  float  f32x4;
typedef __attribute__((ext_vector_type(4)))  __bf16 bf16x4;

union FragBf {
    v16bf v;
    u32x4 u[2];
};

// ---------- CDNA5 async global->LDS path (guarded; sync fallback) ----------
#if defined(__has_builtin)
#  if __has_builtin(__builtin_amdgcn_global_load_async_to_lds_b128) && \
      __has_builtin(__builtin_amdgcn_s_wait_asynccnt)
#    define HAVE_ASYNC_LDS 1
#  endif
#endif
#ifndef HAVE_ASYNC_LDS
#  define HAVE_ASYNC_LDS 0
#endif

#if HAVE_ASYNC_LDS
// signature (from clang diagnostics): (i32x4 AS1* gsrc, i32x4 AS3* ldst, imm offset, imm cpol)
#define ASYNC_COPY16(gp, lp)                                                   \
    __builtin_amdgcn_global_load_async_to_lds_b128(                            \
        (__attribute__((address_space(1))) i32x4*)(gp),                        \
        (__attribute__((address_space(3))) i32x4*)(lp), 0, 0)
#define WAIT_ASYNC(n) __builtin_amdgcn_s_wait_asynccnt(n)
#else
#define ASYNC_COPY16(gp, lp) (*(u32x4*)(lp) = *(const u32x4*)(gp))
#define WAIT_ASYNC(n)
#endif

// ---------- fp32 -> bf16 convert ----------
__global__ __launch_bounds__(256)
void cvt_f32_to_bf16(const float* __restrict__ x, __bf16* __restrict__ y, long n)
{
    long i = ((long)blockIdx.x * 256 + threadIdx.x) * 4;
    if (i + 3 < n) {
        f32x4 a = *(const f32x4*)(x + i);
        bf16x4 o;
        o.x = (__bf16)a.x; o.y = (__bf16)a.y;
        o.z = (__bf16)a.z; o.w = (__bf16)a.w;
        *(bf16x4*)(y + i) = o;
    }
}

// ---------- bf16 32x32-tiled transpose: dst[c][r] = src[r][c] ----------
__global__ __launch_bounds__(256)
void transpose_bf16(const __bf16* __restrict__ src, __bf16* __restrict__ dst,
                    int rows, int cols)
{
    __shared__ __bf16 tile[32][33];
    const int r0 = blockIdx.x * 32;
    const int c0 = blockIdx.y * 32;
    const long zoff = (long)blockIdx.z * (long)rows * (long)cols;
    const int tx = threadIdx.x & 31;
    const int ty = threadIdx.x >> 5;   // 0..7
#pragma unroll
    for (int j = 0; j < 4; ++j)
        tile[ty + j * 8][tx] = src[zoff + (long)(r0 + ty + j * 8) * cols + (c0 + tx)];
    __syncthreads();
#pragma unroll
    for (int j = 0; j < 4; ++j)
        dst[zoff + (long)(c0 + ty + j * 8) * rows + (r0 + tx)] = tile[tx][ty + j * 8];
}

// ---------- LDS tile geometry ----------
// 128 rows x 32 bf16 per K-step; rows padded to 40 elements (80B) so that a
// wave32 ds_load_b128 with row stride 20 banks tiles all 64 banks conflict-free.
#define LDS_ROW   40
#define A_ELEMS   (128 * LDS_ROW)          // 5120 elems = 10KB
#define TILE_ELEMS (2 * A_ELEMS)           // A tile + B tile per buffer

__device__ __forceinline__ v16bf ld_fragA(const __bf16* base, int row, int hi)
{
    FragBf f;
    f.u[0] = *(const u32x4*)(base + row * LDS_ROW + hi * 8);
    f.u[1] = *(const u32x4*)(base + row * LDS_ROW + 16 + hi * 8);
    return f.v;
}

__device__ __forceinline__ v16bf ld_fragB(const __bf16* base, int row, int hi)
{
    FragBf f;
    f.u[0] = *(const u32x4*)(base + row * LDS_ROW + hi * 8);
    f.u[1] = *(const u32x4*)(base + row * LDS_ROW + 16 + hi * 8);
    return f.v;
}

// Issue one K-stage of async copies: 128x32 A tile + 128x32 B tile.
// 512 16B-chunks each; 256 threads x 2 chunks; per wave: 4 async instructions.
__device__ __forceinline__ void stage_copy(const __bf16* __restrict__ Ag,
                                           const __bf16* __restrict__ Bg,
                                           __bf16* lA, __bf16* lB,
                                           int K, int tid)
{
#pragma unroll
    for (int i = 0; i < 2; ++i) {
        const int id = tid + i * 256;      // 0..511
        const int r  = id >> 2;            // 0..127
        const int c  = (id & 3) * 8;       // element offset 0,8,16,24
        const __bf16* ga = Ag + (long)r * K + c;
        const __bf16* gb = Bg + (long)r * K + c;
        ASYNC_COPY16(ga, lA + r * LDS_ROW + c);
        ASYNC_COPY16(gb, lB + r * LDS_ROW + c);
        if (i == 0) {                      // keep L2 ahead by 2 stages
            __builtin_prefetch(ga + 64, 0, 1);
            __builtin_prefetch(gb + 64, 0, 1);
        }
    }
}

// ---------- WMMA GEMM: C[m][n] = sum_k A[m][k] * Bw[n][k]  (+bias / *scale+mask) ----
// Block: 256 threads = 8 waves (4 m-waves x 2 n-waves). Block tile 128(M) x 128(N).
// Wave tile 32x64 = 2x4 of v_wmma_f32_16x16x32_bf16. Double-buffered LDS staging
// via global_load_async_to_lds_b128 + s_wait_asynccnt.
template<bool OUT_BF16, bool USE_BIAS, bool MASK_SCALE>
__global__ __launch_bounds__(256)
void gemm_wmma_bf16(const __bf16* __restrict__ A,  long batchA,
                    const __bf16* __restrict__ Bw, long batchB,
                    void* __restrict__ Cout,       long batchC,
                    const float* __restrict__ bias,
                    const int* __restrict__ mask,  long batchM,
                    int N, int K, float scale)
{
    __shared__ __attribute__((aligned(16))) __bf16 sm[2][TILE_ELEMS];

    const int tid  = threadIdx.x;
    const int lane = tid & 31;
    const int wave = tid >> 5;            // 0..7
    const int wm   = wave >> 1;           // 0..3  -> 32 rows each
    const int wn   = wave & 1;            // 0..1  -> 64 cols each
    const int ml   = lane & 15;
    const int hi   = lane >> 4;           // half-wave select

    const int z = blockIdx.z;
    A  += (long)z * batchA + (long)(blockIdx.y * 128) * K;
    Bw += (long)z * batchB + (long)(blockIdx.x * 128) * K;

    const int aRow = wm * 32 + ml;        // + mt*16
    const int bRow = wn * 64 + ml;        // + nt*16

    v8f acc[2][4] = {};

    int buf = 0;
    stage_copy(A, Bw, &sm[0][0], &sm[0][A_ELEMS], K, tid);

    for (int k0 = 0; k0 < K; k0 += 32) {
        const bool more = (k0 + 32) < K;
        if (more) {
            stage_copy(A + k0 + 32, Bw + k0 + 32,
                       &sm[buf ^ 1][0], &sm[buf ^ 1][A_ELEMS], K, tid);
            WAIT_ASYNC(4);                 // my writes to current buf landed
        } else {
            WAIT_ASYNC(0);                 // final stage: everything landed
        }
        __syncthreads();                   // everyone's writes landed

        const __bf16* lA = &sm[buf][0];
        const __bf16* lB = &sm[buf][A_ELEMS];

        v16bf a0 = ld_fragA(lA, aRow,      hi);
        v16bf a1 = ld_fragA(lA, aRow + 16, hi);
        v16bf b0 = ld_fragB(lB, bRow,      hi);
        v16bf b1 = ld_fragB(lB, bRow + 16, hi);
        v16bf b2 = ld_fragB(lB, bRow + 32, hi);
        v16bf b3 = ld_fragB(lB, bRow + 48, hi);

        acc[0][0] = __builtin_amdgcn_wmma_f32_16x16x32_bf16(false, a0, false, b0, (short)0, acc[0][0], false, false);
        acc[0][1] = __builtin_amdgcn_wmma_f32_16x16x32_bf16(false, a0, false, b1, (short)0, acc[0][1], false, false);
        acc[0][2] = __builtin_amdgcn_wmma_f32_16x16x32_bf16(false, a0, false, b2, (short)0, acc[0][2], false, false);
        acc[0][3] = __builtin_amdgcn_wmma_f32_16x16x32_bf16(false, a0, false, b3, (short)0, acc[0][3], false, false);
        acc[1][0] = __builtin_amdgcn_wmma_f32_16x16x32_bf16(false, a1, false, b0, (short)0, acc[1][0], false, false);
        acc[1][1] = __builtin_amdgcn_wmma_f32_16x16x32_bf16(false, a1, false, b1, (short)0, acc[1][1], false, false);
        acc[1][2] = __builtin_amdgcn_wmma_f32_16x16x32_bf16(false, a1, false, b2, (short)0, acc[1][2], false, false);
        acc[1][3] = __builtin_amdgcn_wmma_f32_16x16x32_bf16(false, a1, false, b3, (short)0, acc[1][3], false, false);

        __syncthreads();                   // reads done before buf is overwritten
        buf ^= 1;
    }

    // Epilogue. C/D layout: VGPR r, lane: m = r + hi*8, n = ml.
    const int mBase = blockIdx.y * 128 + wm * 32;
    const int nBase = blockIdx.x * 128 + wn * 64;
#pragma unroll
    for (int mt = 0; mt < 2; ++mt) {
#pragma unroll
        for (int nt = 0; nt < 4; ++nt) {
            const int n = nBase + nt * 16 + ml;
            float bv = 0.0f;
            if (USE_BIAS) bv = bias[n];
#pragma unroll
            for (int r = 0; r < 8; ++r) {
                const int m = mBase + mt * 16 + hi * 8 + r;
                const long idx = (long)m * N + n;
                float vvv = acc[mt][nt][r];
                if (MASK_SCALE) {
                    vvv *= scale;
                    if (mask[(long)z * batchM + idx] == 0) vvv = -1e9f;
                }
                if (USE_BIAS) vvv += bv;
                if (OUT_BF16)
                    ((__bf16*)Cout)[(long)z * batchC + idx] = (__bf16)vvv;
                else
                    ((float*)Cout)[(long)z * batchC + idx] = vvv;
            }
        }
    }
}

// ---------- row softmax over 2048 fp32 -> bf16 ----------
__global__ __launch_bounds__(256)
void softmax_row_2048(const float* __restrict__ scores, __bf16* __restrict__ attn)
{
    const int tid = threadIdx.x;
    const long row = blockIdx.x;
    const float* src = scores + row * 2048;
    __bf16* dst = attn + row * 2048;

    float v[8];
    float mx = -3.4e38f;
#pragma unroll
    for (int i = 0; i < 8; ++i) {
        v[i] = src[tid + i * 256];
        mx = fmaxf(mx, v[i]);
    }
    __shared__ float red[256];
    red[tid] = mx; __syncthreads();
    for (int s = 128; s > 0; s >>= 1) {
        if (tid < s) red[tid] = fmaxf(red[tid], red[tid + s]);
        __syncthreads();
    }
    mx = red[0]; __syncthreads();

    float sum = 0.0f;
#pragma unroll
    for (int i = 0; i < 8; ++i) {
        v[i] = __expf(v[i] - mx);
        sum += v[i];
    }
    red[tid] = sum; __syncthreads();
    for (int s = 128; s > 0; s >>= 1) {
        if (tid < s) red[tid] += red[tid + s];
        __syncthreads();
    }
    const float inv = 1.0f / red[0];
#pragma unroll
    for (int i = 0; i < 8; ++i)
        dst[tid + i * 256] = (__bf16)(v[i] * inv);
}

// ---------- host driver ----------
extern "C" void kernel_launch(void* const* d_in, const int* in_sizes, int n_in,
                              void* d_out, int out_size, void* d_ws, size_t ws_size,
                              hipStream_t stream)
{
    const int  Bb = 4, S = 2048, H = 1024;
    const long SH  = (long)S * H;        // 2M
    const long BSH = (long)Bb * SH;      // 8M
    const long SS  = (long)S * S;        // 4M
    const long HH  = (long)H * H;        // 1M
    const float scale = 0.03125f;        // 1/sqrt(1024)

    const float* query = (const float*)d_in[0];
    const float* key   = (const float*)d_in[1];
    const float* value = (const float*)d_in[2];
    const int*   mask  = (const int*)  d_in[3];
    const float* Wq = (const float*)d_in[4];
    const float* bq = (const float*)d_in[5];
    const float* Wk = (const float*)d_in[6];
    const float* bk = (const float*)d_in[7];
    const float* Wv = (const float*)d_in[8];
    const float* bv = (const float*)d_in[9];
    const float* Wo = (const float*)d_in[10];
    const float* bo = (const float*)d_in[11];
    float* out = (float*)d_out;

    // workspace layout
    char* ws = (char*)d_ws;
    size_t off = 0;
    auto carve = [&](size_t bytes) -> char* {
        char* p = ws + off;
        off += (bytes + 255) & ~(size_t)255;
        return p;
    };
    __bf16* xq  = (__bf16*)carve(BSH * 2);
    __bf16* xk  = (__bf16*)carve(BSH * 2);
    __bf16* xv  = (__bf16*)carve(BSH * 2);
    __bf16* wqb = (__bf16*)carve(HH * 2);
    __bf16* wkb = (__bf16*)carve(HH * 2);
    __bf16* wvb = (__bf16*)carve(HH * 2);
    __bf16* wob = (__bf16*)carve(HH * 2);
    __bf16* qb  = (__bf16*)carve(BSH * 2);
    __bf16* kb  = (__bf16*)carve(BSH * 2);
    __bf16* vb  = (__bf16*)carve(BSH * 2);
    __bf16* vT  = (__bf16*)carve(BSH * 2);   // per batch: [H][S]
    float*  sc  = (float*) carve((size_t)Bb * SS * 4);
    __bf16* at  = (__bf16*)carve((size_t)Bb * SS * 2);
    __bf16* cx  = (__bf16*)carve(BSH * 2);
    (void)ws_size; (void)n_in; (void)in_sizes; (void)out_size;

    // 1) convert inputs + weights to bf16
    cvt_f32_to_bf16<<<(int)(BSH / 1024), 256, 0, stream>>>(query, xq, BSH);
    cvt_f32_to_bf16<<<(int)(BSH / 1024), 256, 0, stream>>>(key,   xk, BSH);
    cvt_f32_to_bf16<<<(int)(BSH / 1024), 256, 0, stream>>>(value, xv, BSH);
    cvt_f32_to_bf16<<<(int)(HH  / 1024), 256, 0, stream>>>(Wq, wqb, HH);
    cvt_f32_to_bf16<<<(int)(HH  / 1024), 256, 0, stream>>>(Wk, wkb, HH);
    cvt_f32_to_bf16<<<(int)(HH  / 1024), 256, 0, stream>>>(Wv, wvb, HH);
    cvt_f32_to_bf16<<<(int)(HH  / 1024), 256, 0, stream>>>(Wo, wob, HH);

    // 2) projections: [8192,1024] x Wt[1024,1024] + b  -> bf16
    {
        dim3 g(H / 128, (Bb * S) / 128, 1);
        gemm_wmma_bf16<true, true, false><<<g, 256, 0, stream>>>(
            xq, 0, wqb, 0, qb, 0, bq, nullptr, 0, H, H, 1.0f);
        gemm_wmma_bf16<true, true, false><<<g, 256, 0, stream>>>(
            xk, 0, wkb, 0, kb, 0, bk, nullptr, 0, H, H, 1.0f);
        gemm_wmma_bf16<true, true, false><<<g, 256, 0, stream>>>(
            xv, 0, wvb, 0, vb, 0, bv, nullptr, 0, H, H, 1.0f);
    }

    // 3) v -> vT per batch ([S,H] -> [H,S])
    {
        dim3 g(S / 32, H / 32, Bb);
        transpose_bf16<<<g, 256, 0, stream>>>(vb, vT, S, H);
    }

    // 4) scores = (q . k^T) * scale, masked  -> fp32
    {
        dim3 g(S / 128, S / 128, Bb);
        gemm_wmma_bf16<false, false, true><<<g, 256, 0, stream>>>(
            qb, SH, kb, SH, sc, SS, nullptr, mask, SS, S, H, scale);
    }

    // 5) softmax rows -> bf16 attn
    softmax_row_2048<<<(int)(Bb * S), 256, 0, stream>>>(sc, at);

    // 6) context = attn . v  (B operand = vT) -> bf16
    {
        dim3 g(H / 128, S / 128, Bb);
        gemm_wmma_bf16<true, false, false><<<g, 256, 0, stream>>>(
            at, SS, vT, SH, cx, SH, nullptr, nullptr, 0, H, S, 1.0f);
    }

    // 7) out = context . Wo^T + bo -> fp32
    {
        dim3 g(H / 128, (Bb * S) / 128, 1);
        gemm_wmma_bf16<false, true, false><<<g, 256, 0, stream>>>(
            cx, 0, wob, 0, out, 0, bo, nullptr, 0, H, H, 1.0f);
    }
}
